// INN_Unified_44976897523915
// MI455X (gfx1250) — matrix operand order repaired and checked by reading, not compiled, not run on hardware
//
#include <hip/hip_runtime.h>
#include <hip/hip_bf16.h>
#include <math.h>

typedef __attribute__((ext_vector_type(16))) _Float16 v16h;
typedef __attribute__((ext_vector_type(8)))  float    v8f;

#define DN    128
#define HN    256
#define NN    4
#define VV    100
#define VP    112
#define BB    64
#define SS    2048
#define BT    16          // batch rows per workgroup == WMMA M
#define NWG   (BB/BT)     // 4 persistent workgroups
#define NTHR  256
#define NWAVE 8
#define G3H   768

// ---- f16 weight workspace layout (element offsets) ----
#define OFF_WIH   0
#define SZ_WIH    (NN*G3H*DN)            // 393216
#define OFF_WHH   (OFF_WIH + SZ_WIH)
#define SZ_WHH    (NN*G3H*HN)            // 786432
#define OFF_WOP   (OFF_WHH + SZ_WHH)
#define SZ_WOP    (NN*DN*HN)             // 131072
#define OFF_WQ    (OFF_WOP + SZ_WOP)
#define OFF_WK    (OFF_WQ  + DN*DN)
#define OFF_WV    (OFF_WK  + DN*DN)
#define OFF_WF1   (OFF_WV  + DN*DN)
#define OFF_WF2   (OFF_WF1 + DN*DN)
#define OFF_WWO   (OFF_WF2 + DN*DN)
#define OFF_WFB   (OFF_WWO + DN*DN)
#define OFF_WR    (OFF_WFB + DN*DN)
#define SZ_WR     (VP*(NN*DN))           // 57344
#define F16_TOTAL (OFF_WR + SZ_WR)       // 1,482,752 halves

// ---- f32 workspace region (element offsets within f32 region) ----
#define OFF_BHH   0                      // [4][768] materialized b_hh
#define OFF_BSUM  (OFF_BHH + NN*G3H)     // [4][768] b_ih + b_hh (r/z fused bias)
#define OFF_BRP   (OFF_BSUM + NN*G3H)    // [112] padded br
#define F32_TOTAL (OFF_BRP + VP)

// =====================  WMMA tile helpers  =====================
// A tile (16x32 f16) from f16 row-major source. ISA 7.12.2 16-bit A layout:
// lane L holds row M=L%16; two 8-element K chunks at (L/16)*8 and (L/16)*8+16.
__device__ __forceinline__ v16h load_ah(const _Float16* A, int lda, int m0, int k0, int lane) {
  const _Float16* p = A + (m0 + (lane & 15)) * lda + k0 + ((lane >> 4) << 3);
  v16h a;
#pragma unroll
  for (int i = 0; i < 8; ++i) { a[i] = p[i]; a[i + 8] = p[i + 16]; }
  return a;
}
// B tile (32x16 f16) from row-major W[ncols][K] (computing A @ W^T).
// Lane L holds column n=L%16; two 8-element K chunks at (L/16)*8 and (L/16)*8+16.
__device__ __forceinline__ v16h load_bh(const _Float16* W, int ldw, int n0, int k0, int lane) {
  const _Float16* p = W + (n0 + (lane & 15)) * ldw + k0 + ((lane >> 4) << 3);
  v16h b;
#pragma unroll
  for (int i = 0; i < 8; ++i) { b[i] = p[i]; b[i + 8] = p[i + 16]; }
  return b;
}

// D = A[MxK] @ W^T (+ optional A2[MxK2] @ W2^T) + bias.
// A tiles cached in VGPRs across the N-tile loop (K,K2 compile-time => full unroll).
// Optional f32 and/or f16 destinations. Tiles distributed across the 8 waves.
template <int K, int K2>
__device__ __forceinline__ void gemm(const _Float16* A, int lda, const _Float16* W, int ldw,
                                     const _Float16* A2, int lda2, const _Float16* W2, int ldw2,
                                     const float* bias, float* D32, _Float16* D16, int ldd,
                                     int M, int Ncols, int wave, int lane) {
  constexpr int KT = K / 32;
  constexpr int KT2 = K2 / 32;
  const int nt = Ncols >> 4;
  const int mt = M >> 4;
  for (int mi = 0; mi < mt; ++mi) {
    const int m0 = mi << 4;
    v16h ac[KT + KT2 + 1];
#pragma unroll
    for (int kk = 0; kk < KT; ++kk) ac[kk] = load_ah(A, lda, m0, kk * 32, lane);
    if constexpr (K2 > 0) {
#pragma unroll
      for (int kk = 0; kk < KT2; ++kk) ac[KT + kk] = load_ah(A2, lda2, m0, kk * 32, lane);
    }
    for (int ni = wave; ni < nt; ni += NWAVE) {
      const int n0 = ni << 4;
      v8f acc;
      const float bvv = bias[n0 + (lane & 15)];
#pragma unroll
      for (int i = 0; i < 8; ++i) acc[i] = bvv;
#pragma unroll
      for (int kk = 0; kk < KT; ++kk) {
        v16h b = load_bh(W, ldw, n0, kk * 32, lane);
        acc = __builtin_amdgcn_wmma_f32_16x16x32_f16(false, ac[kk], false, b, (short)0, acc, false, false);
      }
      if constexpr (K2 > 0) {
#pragma unroll
        for (int kk = 0; kk < KT2; ++kk) {
          v16h b = load_bh(W2, ldw2, n0, kk * 32, lane);
          acc = __builtin_amdgcn_wmma_f32_16x16x32_f16(false, ac[KT + kk], false, b, (short)0, acc, false, false);
        }
      }
      // D layout: VGPR v, lane l -> row m0 + v + 8*(l/16), col n0 + l%16
      const int rbase = (m0 + ((lane >> 4) << 3)) * ldd + n0 + (lane & 15);
      if (D32) {
        float* dp = D32 + rbase;
#pragma unroll
        for (int i = 0; i < 8; ++i) dp[i * ldd] = acc[i];
      }
      if (D16) {
        _Float16* dp = D16 + rbase;
#pragma unroll
        for (int i = 0; i < 8; ++i) dp[i * ldd] = (_Float16)acc[i];
      }
    }
  }
}

// =====================  prep: fp32 -> f16 weights, materialize locked GRU  =====================
__global__ void prep_kernel(const float* W_ih, const float* W_hh_c, const float* W_op,
                            const float* Wq, const float* Wk, const float* Wv,
                            const float* Wf1, const float* Wf2, const float* Wwo,
                            const float* Wfb, const float* Wr,
                            const float* b_ih, const float* b_hh_c, const float* br,
                            _Float16* wsH, float* wsF) {
  long idx = (long)blockIdx.x * blockDim.x + threadIdx.x;
  const long r0 = SZ_WIH;
  const long r1 = r0 + SZ_WHH;
  const long r2 = r1 + SZ_WOP;
  const long r3 = r2 + 7L * DN * DN;
  const long r4 = r3 + SZ_WR;
  const long r5 = r4 + NN * G3H;
  const long r6 = r5 + NN * G3H;
  const long r7 = r6 + VP;
  if (idx < r0) {
    wsH[OFF_WIH + idx] = (_Float16)W_ih[idx];
  } else if (idx < r1) {
    long i = idx - r0;
    int n = (int)(i / (G3H * HN));
    long rem = i - (long)n * (G3H * HN);
    int row = (int)(rem / HN), col = (int)(rem % HN);
    float v;
    if (n < 2) v = (row >= 2 * HN && col == row - 2 * HN) ? 1.f : 0.f;  // locked: n-gate rows = I
    else       v = W_hh_c[(long)(n - 2) * (G3H * HN) + rem];
    wsH[OFF_WHH + i] = (_Float16)v;
  } else if (idx < r2) {
    long i = idx - r1;
    wsH[OFF_WOP + i] = (_Float16)W_op[i];
  } else if (idx < r3) {
    long i = idx - r2;
    int m = (int)(i / (DN * DN));
    long j = i - (long)m * (DN * DN);
    const float* Ms[7] = {Wq, Wk, Wv, Wf1, Wf2, Wwo, Wfb};
    wsH[OFF_WQ + i] = (_Float16)Ms[m][j];
  } else if (idx < r4) {
    long i = idx - r3;
    int row = (int)(i / (NN * DN)), col = (int)(i % (NN * DN));
    wsH[OFF_WR + i] = (_Float16)(row < VV ? Wr[(long)row * (NN * DN) + col] : 0.f);
  } else if (idx < r5) {
    long i = idx - r4;
    int n = (int)(i / G3H), col = (int)(i % G3H);
    float v;
    if (n < 2) v = (col >= HN && col < 2 * HN) ? 5.f : 0.f;             // locked: z-gate bias = 5
    else       v = b_hh_c[(n - 2) * G3H + col];
    wsF[OFF_BHH + i] = v;
  } else if (idx < r6) {
    long i = idx - r5;
    int n = (int)(i / G3H), col = (int)(i % G3H);
    float v;
    if (n < 2) v = (col >= HN && col < 2 * HN) ? 5.f : 0.f;
    else       v = b_hh_c[(n - 2) * G3H + col];
    wsF[OFF_BSUM + i] = v + b_ih[i];                                    // fused r/z bias
  } else if (idx < r7) {
    long i = idx - r6;
    wsF[OFF_BRP + i] = (i < VV) ? br[i] : 0.f;
  }
}

// =====================  persistent recurrent kernel: 1 WG = 16 batch rows, 2048 steps =====================
__global__ __launch_bounds__(NTHR, 1) void step_kernel(
    const int* __restrict__ x, const float* __restrict__ emb,
    const float* __restrict__ b_ih, const float* __restrict__ b_op,
    const float* __restrict__ bq, const float* __restrict__ bk, const float* __restrict__ bv_,
    const float* __restrict__ bf1, const float* __restrict__ bf2, const float* __restrict__ bwo,
    const float* __restrict__ gamma, const float* __restrict__ beta, const float* __restrict__ bfb,
    const _Float16* __restrict__ wsH, const float* __restrict__ wsF,
    float* __restrict__ out) {
  __shared__ float sh_h[NN][BT][HN];                        // f32 GRU state master (64KB)
  __shared__ __align__(16) _Float16 sh_h16[NN][BT][HN];     // f16 mirror for WMMA A (32KB)
  __shared__ __align__(16) _Float16 fb16[BT][DN];           // workspace-vector carry (f16)
  __shared__ float sh_g[BT][G3H];                           // gates [r|z|gi_n] (48KB)
  __shared__ float sh_gn[BT][HN];                           // gh_n (16KB); aliased by attn scores
  __shared__ __align__(16) _Float16 c16[BT][DN];            // feedback-injected input (f16)
  __shared__ __align__(16) _Float16 o16[BT * NN][DN];       // out_proj rows b*4+n (f16, 16KB)
  __shared__ float sA[BT * NN][DN];                         // f32 scratch (q / pre-gelu / logits)
  __shared__ float sB[BT * NN][DN];                         // f32 scratch (k / res)
  __shared__ float sC[BT * NN][DN];                         // f32 scratch (v)
  __shared__ __align__(16) _Float16 sM16[BT * NN][DN];      // mixed, then proc (f16)
  __shared__ __align__(16) _Float16 sP16[BT * NN][DN];      // proc1, then outn (f16)
  // total: 319,488 B < 320 KB

  const int tid = threadIdx.x, lane = tid & 31, wave = tid >> 5;
  const int bBase = blockIdx.x * BT;
  const float* bhh = wsF + OFF_BHH;
  const float* bsum = wsF + OFF_BSUM;
  const float* brp = wsF + OFF_BRP;
  const float inv_sqrt_d = 0.08838834764831845f;  // 1/sqrt(128)
  float* scb = &sh_gn[0][0];                      // attn score alias (dead region in attn phase)

  for (int i = tid; i < NN * BT * HN; i += NTHR) { ((float*)sh_h)[i] = 0.f; ((_Float16*)sh_h16)[i] = (_Float16)0.f; }
  for (int i = tid; i < BT * DN; i += NTHR) fb16[0][i] = (_Float16)0.f;
  __syncthreads();

#pragma unroll 1
  for (int s = 0; s < SS; ++s) {
    // c_pre = fb @ Wfb^T + bfb -> sA[:16]
    gemm<DN, 0>(&fb16[0][0], DN, wsH + OFF_WFB, DN, nullptr, 0, nullptr, 0,
                bfb, &sA[0][0], nullptr, DN, BT, DN, wave, lane);
    __syncthreads();
    for (int i = tid; i < BT * DN; i += NTHR) {
      int b = i >> 7, d = i & 127;
      int tok = x[(bBase + b) * SS + s];
      c16[b][d] = (_Float16)(sA[b][d] + emb[tok * DN + d]);
    }
    __syncthreads();

#pragma unroll 1
    for (int n = 0; n < NN; ++n) {
      const _Float16* Wih_n = wsH + OFF_WIH + n * G3H * DN;
      const _Float16* Whh_n = wsH + OFF_WHH + n * G3H * HN;
      __builtin_prefetch(Whh_n, 0, 1);
      // r,z fused: g[:, :512] = c@Wih^T + h@Whh^T + (b_ih+b_hh)
      gemm<DN, HN>(&c16[0][0], DN, Wih_n, DN,
                   &sh_h16[n][0][0], HN, Whh_n, HN,
                   bsum + n * G3H, &sh_g[0][0], nullptr, G3H, BT, 2 * HN, wave, lane);
      // n-gate input half:  g[:, 512:] = c@Wih[512:]^T + b_ih[512:]
      gemm<DN, 0>(&c16[0][0], DN, Wih_n + 2 * HN * DN, DN, nullptr, 0, nullptr, 0,
                  b_ih + n * G3H + 2 * HN, &sh_g[0][2 * HN], nullptr, G3H, BT, HN, wave, lane);
      // n-gate hidden half: gn = h@Whh[512:]^T + b_hh[512:]
      gemm<HN, 0>(&sh_h16[n][0][0], HN, Whh_n + 2 * HN * HN, HN, nullptr, 0, nullptr, 0,
                  bhh + n * G3H + 2 * HN, &sh_gn[0][0], nullptr, HN, BT, HN, wave, lane);
      __syncthreads();
      for (int i = tid; i < BT * HN; i += NTHR) {
        int b = i >> 8, j = i & 255;
        float r = 1.f / (1.f + expf(-sh_g[b][j]));
        float z = 1.f / (1.f + expf(-sh_g[b][HN + j]));
        float nv = tanhf(sh_g[b][2 * HN + j] + r * sh_gn[b][j]);
        float hn = (1.f - z) * nv + z * sh_h[n][b][j];
        sh_h[n][b][j] = hn;
        sh_h16[n][b][j] = (_Float16)hn;
      }
      __syncthreads();
      // o[b][n][:] = h_n @ W_op[n]^T + b_op[n]  (f16 only; row b*4+n => base +n*DN, ldd 512)
      gemm<HN, 0>(&sh_h16[n][0][0], HN, wsH + OFF_WOP + n * DN * HN, HN, nullptr, 0, nullptr, 0,
                  b_op + n * DN, nullptr, &o16[0][0] + n * DN, NN * DN, BT, DN, wave, lane);
      __syncthreads();
    }

    // q,k,v over the 64 (b,n) rows (f32 outputs; consumed by VALU attention)
    gemm<DN, 0>(&o16[0][0], DN, wsH + OFF_WQ, DN, nullptr, 0, nullptr, 0, bq,  &sA[0][0], nullptr, DN, BT * NN, DN, wave, lane);
    gemm<DN, 0>(&o16[0][0], DN, wsH + OFF_WK, DN, nullptr, 0, nullptr, 0, bk,  &sB[0][0], nullptr, DN, BT * NN, DN, wave, lane);
    gemm<DN, 0>(&o16[0][0], DN, wsH + OFF_WV, DN, nullptr, 0, nullptr, 0, bv_, &sC[0][0], nullptr, DN, BT * NN, DN, wave, lane);
    __syncthreads();
    // scores (N=4: VALU, one (b,i,j) per thread); scb aliases dead sh_gn
    {
      int b = tid >> 4, ii = (tid >> 2) & 3, jj = tid & 3;
      const float* qr = sA[b * 4 + ii];
      const float* kr = sB[b * 4 + jj];
      float sacc = 0.f;
      for (int d = 0; d < DN; ++d) sacc += qr[d] * kr[d];
      scb[b * 16 + ii * 4 + jj] = sacc * inv_sqrt_d;
    }
    __syncthreads();
    if (tid < BT * NN) {
      float* sc = scb + (tid >> 2) * 16 + (tid & 3) * 4;
      float mx = fmaxf(fmaxf(sc[0], sc[1]), fmaxf(sc[2], sc[3]));
      float e0 = expf(sc[0] - mx), e1 = expf(sc[1] - mx), e2 = expf(sc[2] - mx), e3 = expf(sc[3] - mx);
      float inv = 1.f / (e0 + e1 + e2 + e3);
      sc[0] = e0 * inv; sc[1] = e1 * inv; sc[2] = e2 * inv; sc[3] = e3 * inv;
    }
    __syncthreads();
    // mixed = attn @ v -> f16 directly
    for (int i = tid; i < BT * NN * DN; i += NTHR) {
      int r = i >> 7, d = i & 127, b = r >> 2, ii = r & 3;
      const float* sc = scb + b * 16 + ii * 4;
      sM16[r][d] = (_Float16)(sc[0] * sC[b * 4 + 0][d] + sc[1] * sC[b * 4 + 1][d] +
                              sc[2] * sC[b * 4 + 2][d] + sc[3] * sC[b * 4 + 3][d]);
    }
    __syncthreads();
    // proc1 = gelu(mixed @ Wf1^T + bf1): f32 pre-act in sA, f16 post-act in sP16
    gemm<DN, 0>(&sM16[0][0], DN, wsH + OFF_WF1, DN, nullptr, 0, nullptr, 0, bf1, &sA[0][0], nullptr, DN, BT * NN, DN, wave, lane);
    __syncthreads();
    for (int i = tid; i < BT * NN * DN; i += NTHR) {
      float xg = ((float*)sA)[i];
      ((_Float16*)sP16)[i] = (_Float16)(0.5f * xg * (1.f + erff(xg * 0.70710678118654752f)));
    }
    __syncthreads();
    // proc = proc1 @ Wf2^T + bf2 -> sM16 (f16; mixed is dead)
    gemm<DN, 0>(&sP16[0][0], DN, wsH + OFF_WF2, DN, nullptr, 0, nullptr, 0, bf2, nullptr, &sM16[0][0], DN, BT * NN, DN, wave, lane);
    __syncthreads();
    // gvec -> fb carry (f16); res = proc @ Wwo^T + bwo -> sB (f32)
    for (int i = tid; i < BT * DN; i += NTHR) {
      int b = i >> 7, d = i & 127;
      fb16[b][d] = (_Float16)(0.25f * ((float)sM16[b * 4 + 0][d] + (float)sM16[b * 4 + 1][d] +
                                       (float)sM16[b * 4 + 2][d] + (float)sM16[b * 4 + 3][d]));
    }
    gemm<DN, 0>(&sM16[0][0], DN, wsH + OFF_WWO, DN, nullptr, 0, nullptr, 0, bwo, &sB[0][0], nullptr, DN, BT * NN, DN, wave, lane);
    __syncthreads();
    for (int i = tid; i < BT * NN * DN; i += NTHR) ((float*)sB)[i] += (float)((_Float16*)o16)[i];
    __syncthreads();
    // LayerNorm rows of sB -> f16 outn in sP16 (proc1 dead)
    if (tid < BT * NN) {
      const float* row = sB[tid];
      float mu = 0.f;
      for (int d = 0; d < DN; ++d) mu += row[d];
      mu *= (1.f / DN);
      float var = 0.f;
      for (int d = 0; d < DN; ++d) { float t = row[d] - mu; var += t * t; }
      float rstd = rsqrtf(var * (1.f / DN) + 1e-5f);
      for (int d = 0; d < DN; ++d) sP16[tid][d] = (_Float16)((row[d] - mu) * rstd * gamma[d] + beta[d]);
    }
    __syncthreads();
    // logits: [16,512] @ Wr^T (112x512 padded) -> sA[:16][:112]
    __builtin_prefetch(wsH + OFF_WR, 0, 1);
    gemm<NN * DN, 0>(&sP16[0][0], NN * DN, wsH + OFF_WR, NN * DN, nullptr, 0, nullptr, 0,
                     brp, &sA[0][0], nullptr, DN, BT, VP, wave, lane);
    __syncthreads();
    for (int i = tid; i < BT * VV; i += NTHR) {
      int b = i / VV, v = i - b * VV;
      out[((long)(bBase + b) * SS + s) * VV + v] = sA[b][v];
    }
    __syncthreads();
  }

  // final carries: hF [N,B,H] then fbF [B,D] appended after logits
  const long offH = (long)BB * SS * VV;
  const long offFb = offH + (long)NN * BB * HN;
  for (int i = tid; i < NN * BT * HN; i += NTHR) {
    int n = i / (BT * HN), rem = i - n * (BT * HN), b = rem >> 8, k = rem & 255;
    out[offH + (long)n * BB * HN + (long)(bBase + b) * HN + k] = sh_h[n][b][k];
  }
  for (int i = tid; i < BT * DN; i += NTHR) {
    int b = i >> 7, d = i & 127;
    out[offFb + (long)(bBase + b) * DN + d] = (float)fb16[b][d];
  }
}

extern "C" void kernel_launch(void* const* d_in, const int* in_sizes, int n_in,
                              void* d_out, int out_size, void* d_ws, size_t ws_size,
                              hipStream_t stream) {
  const int*   x      = (const int*)d_in[0];
  const float* emb    = (const float*)d_in[1];
  const float* W_ih   = (const float*)d_in[2];
  const float* b_ih   = (const float*)d_in[3];
  const float* W_hh_c = (const float*)d_in[4];
  const float* b_hh_c = (const float*)d_in[5];
  const float* W_op   = (const float*)d_in[6];
  const float* b_op   = (const float*)d_in[7];
  const float* Wq  = (const float*)d_in[8];  const float* bq  = (const float*)d_in[9];
  const float* Wk  = (const float*)d_in[10]; const float* bk  = (const float*)d_in[11];
  const float* Wv  = (const float*)d_in[12]; const float* bv  = (const float*)d_in[13];
  const float* Wf1 = (const float*)d_in[14]; const float* bf1 = (const float*)d_in[15];
  const float* Wf2 = (const float*)d_in[16]; const float* bf2 = (const float*)d_in[17];
  const float* Wwo = (const float*)d_in[18]; const float* bwo = (const float*)d_in[19];
  const float* gamma = (const float*)d_in[20];
  const float* beta  = (const float*)d_in[21];
  const float* Wr  = (const float*)d_in[22]; const float* br  = (const float*)d_in[23];
  const float* Wfb = (const float*)d_in[24]; const float* bfb = (const float*)d_in[25];

  _Float16* wsH = (_Float16*)d_ws;
  float*    wsF = (float*)((char*)d_ws + (size_t)F16_TOTAL * sizeof(_Float16));

  const long prep_total = (long)SZ_WIH + SZ_WHH + SZ_WOP + 7L * DN * DN + SZ_WR +
                          2L * NN * G3H + VP;
  const int prep_blocks = (int)((prep_total + NTHR - 1) / NTHR);
  prep_kernel<<<prep_blocks, NTHR, 0, stream>>>(W_ih, W_hh_c, W_op, Wq, Wk, Wv, Wf1, Wf2,
                                                Wwo, Wfb, Wr, b_ih, b_hh_c, br, wsH, wsF);
  step_kernel<<<NWG, NTHR, 0, stream>>>(x, emb, b_ih, b_op, bq, bk, bv, bf1, bf2, bwo,
                                        gamma, beta, bfb, wsH, wsF, (float*)d_out);
}
